// HSIC_76493367542784
// MI455X (gfx1250) — compile-verified, matrix operand order checked by reading, not run on hardware
//
#include <hip/hip_runtime.h>

typedef float    v2f  __attribute__((ext_vector_type(2)));
typedef float    v8f  __attribute__((ext_vector_type(8)));
typedef _Float16 v8h  __attribute__((ext_vector_type(8)));
typedef _Float16 v16h __attribute__((ext_vector_type(16)));

#define M_PTS 8192
#define KX 128
#define KY 16
#define NMAC 256   // 8192 / 32 macro-tiles per dimension

// Workspace layout (bytes), total ~4.33 MB:
//   [0]        double   sumKL
//   [64]       float    sqx[M]
//   +4*M       float    sqy[M]
//   +4*M       float    rowK[M]
//   +4*M       float    rowL[M]
//   [131328]   _Float16 xh[M*KX]   (hi f16 split of x)
//   +2*M*KX    _Float16 xl[M*KX]   (lo f16 split: x - xh)

static __device__ __forceinline__ v8f wmma_f32(v2f a, v2f b, v8f c) {
    return __builtin_amdgcn_wmma_f32_16x16x4_f32(false, a, false, b,
                                                 (short)0, c, false, false);
}
static __device__ __forceinline__ v8f wmma16(v16h a, v16h b, v8f c) {
    return __builtin_amdgcn_wmma_f32_16x16x32_f16(false, a, false, b,
                                                  (short)0, c, false, false);
}
// A/B fragment for 16-bit 16x32 WMMA: lane's halves are two contiguous
// 8-element groups (K = 8*half+0..7 and K = 16+8*half+0..7) -> 2x b128 loads.
static __device__ __forceinline__ v16h ldfrag16(const _Float16* p) {
    v8h u = *(const v8h*)(p);
    v8h w = *(const v8h*)(p + 16);
    return __builtin_shufflevector(u, w, 0, 1, 2, 3, 4, 5, 6, 7,
                                         8, 9, 10, 11, 12, 13, 14, 15);
}

// ---------------------------------------------------------------------------
// Kernel 1a: squared row norms (f32) + zero accumulators (ws is poisoned and
// never re-zeroed between graph replays, so reinitialize every launch).
// ---------------------------------------------------------------------------
__global__ __launch_bounds__(256) void hsic_prep(const float* __restrict__ x,
                                                 const float* __restrict__ y,
                                                 float* __restrict__ sqx,
                                                 float* __restrict__ sqy,
                                                 float* __restrict__ rowK,
                                                 float* __restrict__ rowL,
                                                 double* __restrict__ sumKL) {
    int i = blockIdx.x * 256 + threadIdx.x;
    if (i >= M_PTS) return;
    float sx = 0.f;
    const float* xr = x + (size_t)i * KX;
    #pragma unroll 8
    for (int k = 0; k < KX; ++k) { float v = xr[k]; sx += v * v; }
    float sy = 0.f;
    const float* yr = y + (size_t)i * KY;
    #pragma unroll
    for (int k = 0; k < KY; ++k) { float v = yr[k]; sy += v * v; }
    sqx[i] = sx;
    sqy[i] = sy;
    rowK[i] = 0.f;
    rowL[i] = 0.f;
    if (i == 0) *sumKL = 0.0;
}

// ---------------------------------------------------------------------------
// Kernel 1b: split x into hi/lo f16 (x = xh + xl, ~22 effective mantissa bits
// for the split-GEMM  x.x^T = xh.xh^T + xh.xl^T + xl.xh^T  (+O(2^-22)).
// ---------------------------------------------------------------------------
__global__ __launch_bounds__(256) void hsic_cvt(const float* __restrict__ x,
                                                _Float16* __restrict__ xh,
                                                _Float16* __restrict__ xl) {
    int e = blockIdx.x * 256 + threadIdx.x;   // M*KX / 256 blocks, coalesced
    float v = x[e];
    _Float16 h = (_Float16)v;
    xh[e] = h;
    xl[e] = (_Float16)(v - (float)h);
}

// ---------------------------------------------------------------------------
// Kernel 2: fused triangular tile pipeline.
//   One wave per 32x32 macro-tile (2x2 blocking of 16x16 WMMA tiles).
//   x-Gram via split-f16 V_WMMA_F32_16X16X32_F16 (3 chained terms into one
//   f32 accumulator); y-Gram (k=16) via V_WMMA_F32_16X16X4_F32.
//   Cyclic pair mapping jm=(im+dj)%256 covers each unordered macro pair once;
//   off-diagonal tiles count twice for sum(K.L) and scatter both column sums
//   and row sums (mirror tile's column sums).  K and L never touch HBM.
// ---------------------------------------------------------------------------
__global__ __launch_bounds__(256) void hsic_tiles(const _Float16* __restrict__ xh,
                                                  const _Float16* __restrict__ xl,
                                                  const float* __restrict__ y,
                                                  const float* __restrict__ sqx,
                                                  const float* __restrict__ sqy,
                                                  float* __restrict__ rowK,
                                                  float* __restrict__ rowL,
                                                  double* __restrict__ sumKL) {
    const int lane = threadIdx.x & 31;
    const int wave = threadIdx.x >> 5;
    const int l16  = lane & 15;
    const int half = lane >> 4;

    const int im = blockIdx.x * 8 + wave;   // 0..255 (uniform per wave)
    const int dj = blockIdx.y;              // 0..128
    if (dj == 128 && im >= 128) return;     // dj==128 pairs appear twice; keep im<128
    const int jm = (im + dj) & (NMAC - 1);
    const int i0 = im * 32;
    const int j0 = jm * 32;

    // ---- Gram macro-tile of x (k = 128): split-f16, 2x2 blocking ----
    v8f cK[2][2] = {};
    {
        const _Float16* ah = xh + (size_t)(i0 + l16) * KX + half * 8;
        const _Float16* al = xl + (size_t)(i0 + l16) * KX + half * 8;
        const _Float16* bh = xh + (size_t)(j0 + l16) * KX + half * 8;
        const _Float16* bl = xl + (size_t)(j0 + l16) * KX + half * 8;
        #pragma unroll
        for (int kc = 0; kc < KX; kc += 32) {
            v16h hA0 = ldfrag16(ah + kc);
            v16h hA1 = ldfrag16(ah + kc + 16 * KX);
            v16h lA0 = ldfrag16(al + kc);
            v16h lA1 = ldfrag16(al + kc + 16 * KX);
            v16h hB0 = ldfrag16(bh + kc);
            v16h hB1 = ldfrag16(bh + kc + 16 * KX);
            v16h lB0 = ldfrag16(bl + kc);
            v16h lB1 = ldfrag16(bl + kc + 16 * KX);

            cK[0][0] = wmma16(hA0, hB0, cK[0][0]);
            cK[0][0] = wmma16(hA0, lB0, cK[0][0]);
            cK[0][0] = wmma16(lA0, hB0, cK[0][0]);

            cK[0][1] = wmma16(hA0, hB1, cK[0][1]);
            cK[0][1] = wmma16(hA0, lB1, cK[0][1]);
            cK[0][1] = wmma16(lA0, hB1, cK[0][1]);

            cK[1][0] = wmma16(hA1, hB0, cK[1][0]);
            cK[1][0] = wmma16(hA1, lB0, cK[1][0]);
            cK[1][0] = wmma16(lA1, hB0, cK[1][0]);

            cK[1][1] = wmma16(hA1, hB1, cK[1][1]);
            cK[1][1] = wmma16(hA1, lB1, cK[1][1]);
            cK[1][1] = wmma16(lA1, hB1, cK[1][1]);
        }
    }

    // ---- Gram macro-tile of y (k = 16): f32 WMMA ----
    v8f cL[2][2] = {};
    {
        const float* ya0 = y + (size_t)(i0 + l16) * KY + 2 * half;
        const float* ya1 = ya0 + (size_t)16 * KY;
        const float* yb0 = y + (size_t)(j0 + l16) * KY + 2 * half;
        const float* yb1 = yb0 + (size_t)16 * KY;
        #pragma unroll
        for (int kb = 0; kb < KY; kb += 4) {
            v2f a0 = { ya0[kb], ya0[kb + 1] };
            v2f a1 = { ya1[kb], ya1[kb + 1] };
            v2f b0 = { yb0[kb], yb0[kb + 1] };
            v2f b1 = { yb1[kb], yb1[kb + 1] };
            cL[0][0] = wmma_f32(a0, b0, cL[0][0]);
            cL[0][1] = wmma_f32(a0, b1, cL[0][1]);
            cL[1][0] = wmma_f32(a1, b0, cL[1][0]);
            cL[1][1] = wmma_f32(a1, b1, cL[1][1]);
        }
    }

    // C/D layout: register r of sub-tile (ti,tj) holds element
    //   (i = i0 + ti*16 + r + 8*half,  j = j0 + tj*16 + l16).
    const float sxj[2] = { sqx[j0 + l16], sqx[j0 + 16 + l16] };
    const float syj[2] = { sqy[j0 + l16], sqy[j0 + 16 + l16] };

    float pKL = 0.f;
    float colK[2] = { 0.f, 0.f }, colL[2] = { 0.f, 0.f };

    #pragma unroll
    for (int ti = 0; ti < 2; ++ti) {
        #pragma unroll
        for (int r = 0; r < 8; ++r) {
            const int i = i0 + ti * 16 + r + 8 * half;
            const float sxi = sqx[i];
            const float syi = sqy[i];
            float rsk = 0.f, rsl = 0.f;   // this lane's 2-column row partial
            #pragma unroll
            for (int tj = 0; tj < 2; ++tj) {
                // d = ||xi||^2 + ||xj||^2 - 2 xi.xj ;  K = exp(-d / 1.0)
                const float Kv = __expf(2.0f * cK[ti][tj][r] - sxi - sxj[tj]);
                const float Lv = __expf(2.0f * cL[ti][tj][r] - syi - syj[tj]);
                pKL      += Kv * Lv;
                colK[tj] += Kv;          // column sums: free in C layout
                colL[tj] += Lv;
                rsk      += Kv;
                rsl      += Lv;
            }
            if (dj != 0) {
                // Off-diagonal: mirror tile's column sums == our row sums.
                #pragma unroll
                for (int m = 8; m; m >>= 1) {
                    rsk += __shfl_xor(rsk, m, 16);
                    rsl += __shfl_xor(rsl, m, 16);
                }
                if (l16 == 0) {
                    atomicAdd(&rowK[i], rsk);
                    atomicAdd(&rowL[i], rsl);
                }
            }
        }
    }

    // Column sums (rowsum == colsum by symmetry of K, L): combine halves.
    #pragma unroll
    for (int tj = 0; tj < 2; ++tj) {
        float ck = colK[tj] + __shfl_xor(colK[tj], 16, 32);
        float cl = colL[tj] + __shfl_xor(colL[tj], 16, 32);
        if (half == 0) {
            atomicAdd(&rowK[j0 + tj * 16 + l16], ck);
            atomicAdd(&rowL[j0 + tj * 16 + l16], cl);
        }
    }

    // sum(K.L): off-diagonal macro-tiles count twice (mirror not visited).
    #pragma unroll
    for (int m = 16; m; m >>= 1) pKL += __shfl_xor(pKL, m, 32);
    if (lane == 0) {
        const double w = (dj != 0) ? 2.0 : 1.0;
        atomicAdd(sumKL, w * (double)pKL);
    }
}

// ---------------------------------------------------------------------------
// Kernel 3: finalize.  S = sumKL - (2/m) sK.sL + (1/m^2) (sum K)(sum L);
// hsic = S / (m-1)^2
// ---------------------------------------------------------------------------
__global__ __launch_bounds__(256) void hsic_finalize(const float* __restrict__ rowK,
                                                     const float* __restrict__ rowL,
                                                     const double* __restrict__ sumKL,
                                                     float* __restrict__ out) {
    __shared__ double sdot[256], ssk[256], ssl[256];
    double dot = 0.0, sk = 0.0, sl = 0.0;
    for (int i = threadIdx.x; i < M_PTS; i += 256) {
        double a = rowK[i], b = rowL[i];
        dot += a * b;
        sk  += a;
        sl  += b;
    }
    sdot[threadIdx.x] = dot;
    ssk[threadIdx.x]  = sk;
    ssl[threadIdx.x]  = sl;
    __syncthreads();
    for (int s = 128; s > 0; s >>= 1) {
        if (threadIdx.x < s) {
            sdot[threadIdx.x] += sdot[threadIdx.x + s];
            ssk[threadIdx.x]  += ssk[threadIdx.x + s];
            ssl[threadIdx.x]  += ssl[threadIdx.x + s];
        }
        __syncthreads();
    }
    if (threadIdx.x == 0) {
        const double m = (double)M_PTS;
        double S = *sumKL - (2.0 / m) * sdot[0] + (ssk[0] * ssl[0]) / (m * m);
        out[0] = (float)(S / ((m - 1.0) * (m - 1.0)));
    }
}

// ---------------------------------------------------------------------------
extern "C" void kernel_launch(void* const* d_in, const int* in_sizes, int n_in,
                              void* d_out, int out_size, void* d_ws, size_t ws_size,
                              hipStream_t stream) {
    const float* x = (const float*)d_in[0];   // [8192,128]
    const float* y = (const float*)d_in[1];   // [8192,16]
    float* out = (float*)d_out;

    char* base = (char*)d_ws;
    double*   sumKL = (double*)base;
    float*    sqx   = (float*)(base + 64);
    float*    sqy   = sqx + M_PTS;
    float*    rowK  = sqy + M_PTS;
    float*    rowL  = rowK + M_PTS;
    _Float16* xh    = (_Float16*)(base + 131328);            // 256B aligned
    _Float16* xl    = xh + (size_t)M_PTS * KX;

    hsic_prep<<<dim3(M_PTS / 256), dim3(256), 0, stream>>>(x, y, sqx, sqy, rowK, rowL, sumKL);
    hsic_cvt<<<dim3(M_PTS * KX / 256), dim3(256), 0, stream>>>(x, xh, xl);

    // Triangular coverage: im = blockIdx.x*8 + wave (0..255), dj = blockIdx.y
    // (0..128), jm = (im+dj) % 256; dj==128 restricted to im<128.
    hsic_tiles<<<dim3(32, 129), dim3(256), 0, stream>>>(xh, xl, y, sqx, sqy, rowK, rowL, sumKL);

    hsic_finalize<<<dim3(1), dim3(256), 0, stream>>>(rowK, rowL, sumKL, out);
}
